// LinearTemporalCrossAttention_64046552318269
// MI455X (gfx1250) — matrix operand order repaired and checked by reading, not compiled, not run on hardware
//
#include <hip/hip_runtime.h>
#include <hip/hip_bf16.h>
#include <cstdint>

// ---------------------------------------------------------------------------
// Types for CDNA5 WMMA
// ---------------------------------------------------------------------------
typedef _Float16 v8h  __attribute__((ext_vector_type(8)));
typedef _Float16 v16h __attribute__((ext_vector_type(16)));
typedef float    v8f  __attribute__((ext_vector_type(8)));

static __device__ __forceinline__ float silu_f(float x) {
    return x / (1.0f + __expf(-x));
}

// ---------------------------------------------------------------------------
// f32 (K x N) -> f16 transposed (N x K).  Tiled 32x32 via LDS, 256 threads.
// grid: (N/32, K/32)
// ---------------------------------------------------------------------------
__global__ void cvt_transpose_kernel(const float* __restrict__ W, _Float16* __restrict__ Wt,
                                     int K, int N) {
    __shared__ float t[32][33];
    int n0 = blockIdx.x * 32, k0 = blockIdx.y * 32;
    int c = threadIdx.x & 31, r = threadIdx.x >> 5;  // r: 0..7
#pragma unroll
    for (int j = 0; j < 4; ++j) {
        int kk = r + j * 8;
        t[kk][c] = W[(size_t)(k0 + kk) * N + n0 + c];
    }
    __syncthreads();
#pragma unroll
    for (int j = 0; j < 4; ++j) {
        int nn = r + j * 8;
        Wt[(size_t)(n0 + nn) * K + k0 + c] = (_Float16)t[c][nn];
    }
}

// ---------------------------------------------------------------------------
// Row LayerNorm: f32 in -> f16 out. One block (256 threads) per row.
// ---------------------------------------------------------------------------
__global__ void ln_rows_kernel(const float* __restrict__ X, int C,
                               const float* __restrict__ g, const float* __restrict__ b,
                               _Float16* __restrict__ out) {
    int row = blockIdx.x;
    const float* xr = X + (size_t)row * C;
    __shared__ float s1[256], s2[256];
    float a = 0.f, q = 0.f;
    for (int c = threadIdx.x; c < C; c += 256) { float v = xr[c]; a += v; q += v * v; }
    s1[threadIdx.x] = a; s2[threadIdx.x] = q;
    __syncthreads();
    for (int o = 128; o > 0; o >>= 1) {
        if (threadIdx.x < o) { s1[threadIdx.x] += s1[threadIdx.x + o]; s2[threadIdx.x] += s2[threadIdx.x + o]; }
        __syncthreads();
    }
    float mu = s1[0] / (float)C;
    float var = s2[0] / (float)C - mu * mu;
    float rs = rsqrtf(var + 1e-5f);
    _Float16* orow = out + (size_t)row * C;
    for (int c = threadIdx.x; c < C; c += 256)
        orow[c] = (_Float16)((xr[c] - mu) * rs * g[c] + b[c]);
}

// ---------------------------------------------------------------------------
// Stylization: silu( LN(y)*(1+scale)+shift ) -> f16.  One block per (b,t) row.
// ---------------------------------------------------------------------------
__global__ void style_h_kernel(const _Float16* __restrict__ Y,
                               const float* __restrict__ g, const float* __restrict__ b,
                               const float* __restrict__ embout,
                               _Float16* __restrict__ out) {
    int row = blockIdx.x;
    int bb  = row >> 10;  // row / T (T = 1024)
    const _Float16* yr = Y + (size_t)row * 1024;
    __shared__ float s1[256], s2[256];
    float a = 0.f, q = 0.f;
    for (int c = threadIdx.x; c < 1024; c += 256) { float v = (float)yr[c]; a += v; q += v * v; }
    s1[threadIdx.x] = a; s2[threadIdx.x] = q;
    __syncthreads();
    for (int o = 128; o > 0; o >>= 1) {
        if (threadIdx.x < o) { s1[threadIdx.x] += s1[threadIdx.x + o]; s2[threadIdx.x] += s2[threadIdx.x + o]; }
        __syncthreads();
    }
    float mu = s1[0] * (1.0f / 1024.0f);
    float var = s2[0] * (1.0f / 1024.0f) - mu * mu;
    float rs = rsqrtf(var + 1e-5f);
    const float* eo = embout + (size_t)bb * 2048;
    _Float16* orow = out + (size_t)row * 1024;
    for (int c = threadIdx.x; c < 1024; c += 256) {
        float ln = ((float)yr[c] - mu) * rs * g[c] + b[c];
        float v  = ln * (1.0f + eo[c]) + eo[1024 + c];
        orow[c] = (_Float16)silu_f(v);
    }
}

// ---------------------------------------------------------------------------
// silu(emb) elementwise
// ---------------------------------------------------------------------------
__global__ void silu_kernel(const float* __restrict__ X, float* __restrict__ Y, int n) {
    int i = blockIdx.x * 256 + threadIdx.x;
    if (i < n) Y[i] = silu_f(X[i]);
}

// ---------------------------------------------------------------------------
// embout = silu(emb) @ We + be   (32 x 2048 x 2048) — tiny, VALU kernel.
// ---------------------------------------------------------------------------
__global__ void emb_gemm_kernel(const float* __restrict__ SE, const float* __restrict__ We,
                                const float* __restrict__ be, float* __restrict__ EO) {
    int c = blockIdx.x * 256 + threadIdx.x;   // 0..2047
    int bb = blockIdx.y;
    const float* se = SE + (size_t)bb * 2048;
    float acc = be[c];
    for (int k = 0; k < 2048; ++k)
        acc = fmaf(se[k], We[(size_t)k * 2048 + c], acc);
    EO[(size_t)bb * 2048 + c] = acc;
}

// ---------------------------------------------------------------------------
// Softmax of k over N=77 (axis 1).  One thread per (b, d).
// ---------------------------------------------------------------------------
__global__ void softmax_n_kernel(const float* __restrict__ Kin, _Float16* __restrict__ Kout) {
    int idx = blockIdx.x * 256 + threadIdx.x;  // B*D threads
    int bb = idx >> 10, d = idx & 1023;
    const float* base = Kin + (size_t)bb * 77 * 1024 + d;
    float mx = -3.4e38f;
    for (int n = 0; n < 77; ++n) mx = fmaxf(mx, base[(size_t)n * 1024]);
    float s = 0.f;
    for (int n = 0; n < 77; ++n) s += __expf(base[(size_t)n * 1024] - mx);
    float inv = 1.0f / s;
    _Float16* ob = Kout + (size_t)bb * 77 * 1024 + d;
    for (int n = 0; n < 77; ++n)
        ob[(size_t)n * 1024] = (_Float16)(__expf(base[(size_t)n * 1024] - mx) * inv);
}

// ---------------------------------------------------------------------------
// attT[b,h,l,d] = sum_n ksm[b,n,h*64+d] * v[b,n,h*64+l]   (stored TRANSPOSED)
// One block (256 threads) per (b,h).
// ---------------------------------------------------------------------------
__global__ void att_kernel(const _Float16* __restrict__ Ks, const _Float16* __restrict__ V,
                           _Float16* __restrict__ ATTT) {
    int bh = blockIdx.x;
    int bb = bh >> 4, h = bh & 15;
    __shared__ _Float16 kS[77][64];
    __shared__ _Float16 vS[77][64];
    for (int i = threadIdx.x; i < 77 * 64; i += 256) {
        int n = i >> 6, d = i & 63;
        size_t off = ((size_t)(bb * 77 + n)) * 1024 + h * 64 + d;
        kS[n][d] = Ks[off];
        vS[n][d] = V[off];
    }
    __syncthreads();
    for (int o = threadIdx.x; o < 4096; o += 256) {
        int l = o >> 6, d = o & 63;   // output row = l, col = d (transposed)
        float acc = 0.f;
        for (int n = 0; n < 77; ++n)
            acc = fmaf((float)kS[n][d], (float)vS[n][l], acc);
        ATTT[(size_t)bh * 4096 + o] = (_Float16)acc;
    }
}

// ---------------------------------------------------------------------------
// Stage one 64(M) x 64(K) A tile + 128(N) x 64(K) B tile into LDS buffer `buf`.
// Buffer stride 27648B: As 64x72 f16 (9216B) then Bs 128x72 f16 (18432B).
// ---------------------------------------------------------------------------
static __device__ __forceinline__ void gemm_stage(char* smem, int buf,
    const _Float16* __restrict__ A, int lda,
    const _Float16* __restrict__ Wt, int K,
    int m0, int n0, int k0, int M, int tid)
{
    _Float16 (*As)[72] = (_Float16(*)[72])(smem + buf * 27648);
    _Float16 (*Bs)[72] = (_Float16(*)[72])(smem + buf * 27648 + 9216);
#pragma unroll
    for (int i = 0; i < 4; ++i) {
        int chunk = tid + i * 128;            // 0..511
        int r = chunk >> 3, kc = (chunk & 7) * 8;
        int gm = m0 + r;
        v8h av = {0, 0, 0, 0, 0, 0, 0, 0};
        if (gm < M) av = *(const v8h*)(A + (size_t)gm * lda + k0 + kc);
        *(v8h*)&As[r][kc] = av;
    }
#pragma unroll
    for (int i = 0; i < 8; ++i) {
        int chunk = tid + i * 128;            // 0..1023
        int nr = chunk >> 3, kc = (chunk & 7) * 8;
        *(v8h*)&Bs[nr][kc] = *(const v8h*)(Wt + (size_t)(n0 + nr) * K + k0 + kc);
    }
}

// ---------------------------------------------------------------------------
// Generic f16 WMMA GEMM: C(MxN) = A(MxK) @ W(KxN) + bias.
// W supplied PRE-TRANSPOSED as Wt (N x K row-major).
//   MODE 0: f32 out + bias
//   MODE 1: f16 out + bias
//   MODE 2: per-64-col-head softmax(+bias), * keep-mask, f16 out
//   MODE 3: f32 out + bias + residual X
// Block tile 64x128, BLK_K=64, 128 threads = 4 waves; each wave 32x64 (2x4 WMMA).
// Double-buffered LDS: stage tile k+1 while WMMAs consume tile k (1 barrier/iter).
// ---------------------------------------------------------------------------
template <int MODE>
__global__ __launch_bounds__(128)
void gemm_wmma_kernel(const _Float16* __restrict__ A, int lda,
                      const _Float16* __restrict__ Wt,   // N x K, row stride K
                      const float* __restrict__ bias,
                      void* __restrict__ Cout, int ldc,
                      int M, int K,
                      const float* __restrict__ X,
                      const unsigned char* __restrict__ mask) {
    // Two 27648B tile buffers; epilogue aliases base as Cs 64x132 f32 (33792B).
    __shared__ __attribute__((aligned(16))) char smem[55296];
    float (*Cs)[132] = (float(*)[132])smem;

    const int tid = threadIdx.x;
    const int m0 = blockIdx.y * 64;
    const int n0 = blockIdx.x * 128;
    const int wid = tid >> 5, lane = tid & 31;
    const int wm = wid >> 1, wn = wid & 1;
    const int lr = lane & 15;
    const bool hi = lane >= 16;

    const v8f z8 = {0.f, 0.f, 0.f, 0.f, 0.f, 0.f, 0.f, 0.f};
    v8f acc[2][4] = {{z8, z8, z8, z8}, {z8, z8, z8, z8}};

    // Prologue: stage first tile into buffer 0
    gemm_stage(smem, 0, A, lda, Wt, K, m0, n0, 0, M, tid);
    __syncthreads();

    int buf = 0;
    for (int k0 = 0; k0 < K; k0 += 64) {
        // Stage next tile into the other buffer while this one computes
        if (k0 + 64 < K) {
            gemm_stage(smem, buf ^ 1, A, lda, Wt, K, m0, n0, k0 + 64, M, tid);
            if (k0 + 128 < K) {
                __builtin_prefetch(A + (size_t)(m0 + (tid >> 1)) * lda + k0 + 128, 0, 1);
                __builtin_prefetch(Wt + (size_t)(n0 + tid) * K + k0 + 128, 0, 1);
            }
        }

        _Float16 (*As)[72] = (_Float16(*)[72])(smem + buf * 27648);
        _Float16 (*Bs)[72] = (_Float16(*)[72])(smem + buf * 27648 + 9216);
#pragma unroll
        for (int kk = 0; kk < 64; kk += 32) {
            const int klo = kk + (hi ? 8 : 0);
            const int kb  = kk + (hi ? 8 : 0);
            v16h af[2], bf[4];
#pragma unroll
            for (int mt = 0; mt < 2; ++mt) {
                int r = wm * 32 + mt * 16 + lr;
                v8h a0 = *(const v8h*)&As[r][klo];
                v8h a1 = *(const v8h*)&As[r][klo + 16];
                af[mt] = __builtin_shufflevector(a0, a1, 0, 1, 2, 3, 4, 5, 6, 7, 8, 9, 10, 11, 12, 13, 14, 15);
            }
#pragma unroll
            for (int nt = 0; nt < 4; ++nt) {
                int c = wn * 64 + nt * 16 + lr;
                v8h b0 = *(const v8h*)&Bs[c][kb];
                v8h b1 = *(const v8h*)&Bs[c][kb + 16];
                bf[nt] = __builtin_shufflevector(b0, b1, 0, 1, 2, 3, 4, 5, 6, 7, 8, 9, 10, 11, 12, 13, 14, 15);
            }
#pragma unroll
            for (int mt = 0; mt < 2; ++mt)
#pragma unroll
                for (int nt = 0; nt < 4; ++nt)
                    acc[mt][nt] = __builtin_amdgcn_wmma_f32_16x16x32_f16(
                        false, af[mt], false, bf[nt], (short)0, acc[mt][nt], false, false);
        }
        // Single barrier: next-buffer stores visible AND current buffer free to
        // be overwritten next iteration.
        __syncthreads();
        buf ^= 1;
    }

    // Stage accumulators to LDS (aliases tile buffers — loop ended in a barrier)
#pragma unroll
    for (int mt = 0; mt < 2; ++mt)
#pragma unroll
        for (int nt = 0; nt < 4; ++nt)
#pragma unroll
            for (int v = 0; v < 8; ++v) {
                int r = wm * 32 + mt * 16 + v + (hi ? 8 : 0);
                int c = wn * 64 + nt * 16 + lr;
                Cs[r][c] = acc[mt][nt][v];
            }
    __syncthreads();

    if (MODE == 0 || MODE == 3) {
        // float4 stores: 64 x 128 floats = 2048 chunks of 4
        for (int i = 0; i < 16; ++i) {
            int chunk = tid + i * 128;
            int r = chunk >> 5, c4 = (chunk & 31) * 4;
            int gm = m0 + r;
            if (gm >= M) continue;
            size_t o = (size_t)gm * ldc + n0 + c4;
            float4 v;
            v.x = Cs[r][c4 + 0] + bias[n0 + c4 + 0];
            v.y = Cs[r][c4 + 1] + bias[n0 + c4 + 1];
            v.z = Cs[r][c4 + 2] + bias[n0 + c4 + 2];
            v.w = Cs[r][c4 + 3] + bias[n0 + c4 + 3];
            if (MODE == 3) {
                float4 xr = *(const float4*)(X + o);
                v.x += xr.x; v.y += xr.y; v.z += xr.z; v.w += xr.w;
            }
            *(float4*)((float*)Cout + o) = v;
        }
    } else if (MODE == 1) {
        // v8h stores: 64 x 128 halfs = 1024 chunks of 8
        for (int i = 0; i < 8; ++i) {
            int chunk = tid + i * 128;
            int r = chunk >> 4, c8 = (chunk & 15) * 8;
            int gm = m0 + r;
            if (gm >= M) continue;
            v8h v;
#pragma unroll
            for (int j = 0; j < 8; ++j) v[j] = (_Float16)(Cs[r][c8 + j] + bias[n0 + c8 + j]);
            *(v8h*)((_Float16*)Cout + (size_t)gm * ldc + n0 + c8) = v;
        }
    } else {  // MODE 2: softmax per 64-wide head group (tile spans 2 heads), * mask
        if (tid < 64) {
            int r = tid, gm = m0 + r;
            if (gm < M) {
                float keep = mask[gm] ? 0.f : 1.f;
#pragma unroll
                for (int g = 0; g < 2; ++g) {
                    int cb = g * 64;
                    float mx = -3.4e38f;
                    for (int c = 0; c < 64; ++c) {
                        float v = Cs[r][cb + c] + bias[n0 + cb + c];
                        Cs[r][cb + c] = v;
                        mx = fmaxf(mx, v);
                    }
                    float s = 0.f;
                    for (int c = 0; c < 64; ++c) {
                        float e = __expf(Cs[r][cb + c] - mx);
                        Cs[r][cb + c] = e;
                        s += e;
                    }
                    float inv = keep / s;
                    _Float16* orow = (_Float16*)Cout + (size_t)gm * ldc + n0 + cb;
                    for (int c = 0; c < 64; ++c) orow[c] = (_Float16)(Cs[r][cb + c] * inv);
                }
            }
        }
    }
}

// ---------------------------------------------------------------------------
// Batched y = q_sm @ att per (b,h): 64x64 tile, K=64 (two WMMA k-steps).
// att supplied transposed ([l][d]); output staged via LDS for b128 stores.
// grid: (T/64, B*H), 128 threads.
// ---------------------------------------------------------------------------
__global__ __launch_bounds__(128)
void gemm_y_kernel(const _Float16* __restrict__ Q, const _Float16* __restrict__ ATTT,
                   _Float16* __restrict__ Y) {
    __shared__ __attribute__((aligned(16))) char smem[18432];
    _Float16 (*As)[72] = (_Float16(*)[72])smem;            // q tile [t][d]
    _Float16 (*Bs)[72] = (_Float16(*)[72])(smem + 9216);   // attT [l][d]
    _Float16 (*Ys)[72] = (_Float16(*)[72])smem;            // output staging (aliases As)

    const int tid = threadIdx.x;
    const int t0 = blockIdx.x * 64;
    const int bh = blockIdx.y;
    const int bb = bh >> 4, h = bh & 15;
    const int wid = tid >> 5, lane = tid & 31;
    const int wm = wid >> 1, wn = wid & 1;
    const int lr = lane & 15;
    const bool hi = lane >= 16;

    const _Float16* Abase = Q + ((size_t)(bb * 1024 + t0)) * 1024 + h * 64;
    const _Float16* att   = ATTT + (size_t)bh * 4096;
    _Float16* Ybase       = Y + ((size_t)(bb * 1024 + t0)) * 1024 + h * 64;

#pragma unroll
    for (int i = 0; i < 4; ++i) {  // A: 64x64 halfs
        int chunk = tid + i * 128;
        int r = chunk >> 3, kc = (chunk & 7) * 8;
        *(v8h*)&As[r][kc] = *(const v8h*)(Abase + (size_t)r * 1024 + kc);
    }
#pragma unroll
    for (int i = 0; i < 4; ++i) {  // attT rows -> Bs rows (no scatter)
        int chunk = tid + i * 128;
        int lrow = chunk >> 3, dc = (chunk & 7) * 8;
        *(v8h*)&Bs[lrow][dc] = *(const v8h*)(att + (size_t)lrow * 64 + dc);
    }
    __syncthreads();

    const v8f z8 = {0.f, 0.f, 0.f, 0.f, 0.f, 0.f, 0.f, 0.f};
    v8f acc[2][2] = {{z8, z8}, {z8, z8}};

#pragma unroll
    for (int kk = 0; kk < 64; kk += 32) {
        const int klo = kk + (hi ? 8 : 0);
        const int kb  = kk + (hi ? 8 : 0);
        v16h af[2], bf[2];
#pragma unroll
        for (int mt = 0; mt < 2; ++mt) {
            int r = wm * 32 + mt * 16 + lr;
            v8h a0 = *(const v8h*)&As[r][klo];
            v8h a1 = *(const v8h*)&As[r][klo + 16];
            af[mt] = __builtin_shufflevector(a0, a1, 0, 1, 2, 3, 4, 5, 6, 7, 8, 9, 10, 11, 12, 13, 14, 15);
        }
#pragma unroll
        for (int nt = 0; nt < 2; ++nt) {
            int c = wn * 32 + nt * 16 + lr;
            v8h b0 = *(const v8h*)&Bs[c][kb];
            v8h b1 = *(const v8h*)&Bs[c][kb + 16];
            bf[nt] = __builtin_shufflevector(b0, b1, 0, 1, 2, 3, 4, 5, 6, 7, 8, 9, 10, 11, 12, 13, 14, 15);
        }
#pragma unroll
        for (int mt = 0; mt < 2; ++mt)
#pragma unroll
            for (int nt = 0; nt < 2; ++nt)
                acc[mt][nt] = __builtin_amdgcn_wmma_f32_16x16x32_f16(
                    false, af[mt], false, bf[nt], (short)0, acc[mt][nt], false, false);
    }
    __syncthreads();  // all waves done reading As before aliasing it as Ys

#pragma unroll
    for (int mt = 0; mt < 2; ++mt)
#pragma unroll
        for (int nt = 0; nt < 2; ++nt)
#pragma unroll
            for (int v = 0; v < 8; ++v) {
                int r = wm * 32 + mt * 16 + v + (hi ? 8 : 0);
                int c = wn * 32 + nt * 16 + lr;
                Ys[r][c] = (_Float16)acc[mt][nt][v];
            }
    __syncthreads();

#pragma unroll
    for (int i = 0; i < 4; ++i) {  // coalesced b128 output stores
        int chunk = tid + i * 128;
        int r = chunk >> 3, cc = (chunk & 7) * 8;
        *(v8h*)(Ybase + (size_t)r * 1024 + cc) = *(const v8h*)&Ys[r][cc];
    }
}

// ---------------------------------------------------------------------------
// Host launcher
// ---------------------------------------------------------------------------
extern "C" void kernel_launch(void* const* d_in, const int* in_sizes, int n_in,
                              void* d_out, int out_size, void* d_ws, size_t ws_size,
                              hipStream_t stream) {
    (void)in_sizes; (void)n_in; (void)out_size; (void)ws_size;

    constexpr int Bb = 32, T = 1024, D = 1024, Nn = 77, L = 768, Hh = 16, TE = 2048;
    constexpr int MT = Bb * T;    // 32768
    constexpr int MN = Bb * Nn;   // 2464

    const float* x    = (const float*)d_in[0];
    const float* xf   = (const float*)d_in[1];
    const float* emb  = (const float*)d_in[2];
    const unsigned char* mask = (const unsigned char*)d_in[3];
    const float* ln_g = (const float*)d_in[4];
    const float* ln_b = (const float*)d_in[5];
    const float* tln_g = (const float*)d_in[6];
    const float* tln_b = (const float*)d_in[7];
    const float* sln_g = (const float*)d_in[8];
    const float* sln_b = (const float*)d_in[9];
    const float* Wq = (const float*)d_in[10];
    const float* bq = (const float*)d_in[11];
    const float* Wk = (const float*)d_in[12];
    const float* bk = (const float*)d_in[13];
    const float* Wv = (const float*)d_in[14];
    const float* bv = (const float*)d_in[15];
    const float* We = (const float*)d_in[16];
    const float* be = (const float*)d_in[17];
    const float* Wo = (const float*)d_in[18];
    const float* bo = (const float*)d_in[19];
    float* out = (float*)d_out;

    // Carve workspace (256B aligned chunks)
    char* w = (char*)d_ws;
    auto carve = [&](size_t bytes) -> char* {
        char* p = w;
        w += (bytes + 255) & ~(size_t)255;
        return p;
    };
    _Float16* xh     = (_Float16*)carve((size_t)MT * D * 2);   // LN(x) f16; later reused as h
    _Float16* qsm    = (_Float16*)carve((size_t)MT * D * 2);   // softmaxed q f16
    _Float16* ybuf   = (_Float16*)carve((size_t)MT * D * 2);   // y f16
    _Float16* xfh    = (_Float16*)carve((size_t)MN * L * 2);   // LN(xf) f16
    float*    kbuf   = (float*)carve((size_t)MN * D * 4);      // k pre-softmax f32
    _Float16* ksm    = (_Float16*)carve((size_t)MN * D * 2);   // softmaxed k f16
    _Float16* vbuf   = (_Float16*)carve((size_t)MN * D * 2);   // v f16
    _Float16* attb   = (_Float16*)carve((size_t)Bb * Hh * 64 * 64 * 2);  // att TRANSPOSED
    float*    se     = (float*)carve((size_t)Bb * TE * 4);     // silu(emb)
    float*    embout = (float*)carve((size_t)Bb * 2 * D * 4);
    _Float16* WqT    = (_Float16*)carve((size_t)D * D * 2);    // f16, transposed (N x K)
    _Float16* WkT    = (_Float16*)carve((size_t)L * D * 2);
    _Float16* WvT    = (_Float16*)carve((size_t)L * D * 2);
    _Float16* WoT    = (_Float16*)carve((size_t)D * D * 2);

    // 1) Weight conversions to f16 with transpose (N x K layout for WMMA B-stage)
    cvt_transpose_kernel<<<dim3(D / 32, D / 32), 256, 0, stream>>>(Wq, WqT, D, D);
    cvt_transpose_kernel<<<dim3(D / 32, L / 32), 256, 0, stream>>>(Wk, WkT, L, D);
    cvt_transpose_kernel<<<dim3(D / 32, L / 32), 256, 0, stream>>>(Wv, WvT, L, D);
    cvt_transpose_kernel<<<dim3(D / 32, D / 32), 256, 0, stream>>>(Wo, WoT, D, D);

    // 2) LayerNorms -> f16
    ln_rows_kernel<<<MN, 256, 0, stream>>>(xf, L, tln_g, tln_b, xfh);
    ln_rows_kernel<<<MT, 256, 0, stream>>>(x, D, ln_g, ln_b, xh);

    // 3) Stylization emb path (small, VALU)
    silu_kernel<<<(Bb * TE + 255) / 256, 256, 0, stream>>>(emb, se, Bb * TE);
    emb_gemm_kernel<<<dim3(2 * D / 256, Bb), 256, 0, stream>>>(se, We, be, embout);

    // 4) k / v projections (WMMA, K = 768, N-tile 128)
    gemm_wmma_kernel<0><<<dim3(D / 128, (MN + 63) / 64), 128, 0, stream>>>(
        xfh, L, WkT, bk, kbuf, D, MN, L, nullptr, nullptr);
    gemm_wmma_kernel<1><<<dim3(D / 128, (MN + 63) / 64), 128, 0, stream>>>(
        xfh, L, WvT, bv, vbuf, D, MN, L, nullptr, nullptr);

    // 5) q projection with fused head-softmax + padding mask (WMMA, K = 1024)
    gemm_wmma_kernel<2><<<dim3(D / 128, MT / 64), 128, 0, stream>>>(
        xh, D, WqT, bq, qsm, D, MT, D, nullptr, mask);

    // 6) k softmax over N
    softmax_n_kernel<<<Bb * D / 256, 256, 0, stream>>>(kbuf, ksm);

    // 7) attT = (k^T @ v)^T per (b,h)
    att_kernel<<<Bb * Hh, 256, 0, stream>>>(ksm, vbuf, attb);

    // 8) y = q_sm @ att (batched WMMA)
    gemm_y_kernel<<<dim3(T / 64, Bb * Hh), 128, 0, stream>>>(qsm, attb, ybuf);

    // 9) Stylization: h = silu(LN(y)*(1+scale)+shift) -> f16 (reuse xh)
    style_h_kernel<<<MT, 256, 0, stream>>>(ybuf, sln_g, sln_b, embout, xh);

    // 10) out = x + h @ Wo + bo (WMMA with fused residual)
    gemm_wmma_kernel<3><<<dim3(D / 128, MT / 64), 128, 0, stream>>>(
        xh, D, WoT, bo, out, D, MT, D, x, nullptr);
}
